// Speller_14147622273491
// MI455X (gfx1250) — matrix operand order repaired and checked by reading, not compiled, not run on hardware
//
#include <hip/hip_runtime.h>
#include <hip/hip_bf16.h>
#include <math.h>

// ---------------- problem constants ----------------
#define BB 64
#define TLIS 800
#define DLIS 512
#define HH 512
#define EE 256
#define KK 128
#define VV 128
#define CC 64
#define LTGT 201
#define TDEC 200
#define IN1 (EE + VV + HH)   // 896 : [x(384) | h1(512)]
#define IN2 (HH + HH)        // 1024: [h1 | h2]
#define G4H (4 * HH)         // 2048

typedef _Float16 h16;
typedef __attribute__((ext_vector_type(16))) _Float16 v16h;
typedef __attribute__((ext_vector_type(8)))  float    v8f;

// ---------------- WMMA operand loaders (wave32; contiguous k order, not AMD's table order) ----------------
// A 16x32 f16, this code's k order: lane L holds row M=L%16; lanes<16 hold K in {0..15},
// lanes>=16 hold K in {16..31}; element j -> K = (L/16)*16 + j.
__device__ inline v16h load_a_f16(const h16* X, int ld, int row0, int kbase, int lane) {
  int r = lane & 15, hi = lane >> 4;
  const h16* p = X + (size_t)(row0 + r) * ld + kbase + hi * 16;
  v16h a;
#pragma unroll
  for (int j = 0; j < 8; ++j)  a[j] = p[j];
#pragma unroll
  for (int j = 8; j < 16; ++j) a[j] = p[j];
  return a;
}
__device__ inline v16h load_a_f32(const float* X, int ld, int row0, int kbase, int lane) {
  int r = lane & 15, hi = lane >> 4;
  const float* p = X + (size_t)(row0 + r) * ld + kbase + hi * 16;
  v16h a;
#pragma unroll
  for (int j = 0; j < 8; ++j)  a[j] = (h16)p[j];
#pragma unroll
  for (int j = 8; j < 16; ++j) a[j] = (h16)p[j];
  return a;
}
// B-matrix 32x16 f16, pre-packed so lane L reads 16 contiguous halves (32B):
// element j = W^T[kc*32 + (L/16)*16 + j][nt*16 + L%16]
__device__ inline v16h load_b_pk(const h16* Wpk, int KC, int nt, int kc, int lane) {
  const v16h* p = (const v16h*)(Wpk + ((((size_t)nt * KC + kc) * 32 + lane) << 4));
  return *p;
}

__device__ inline float sigf(float x) { return 1.f / (1.f + __expf(-x)); }

// ---------------- weight packing (one-time) ----------------
// Pack W (N x (Ka+Kb) virtual column-concat of Wa,Wb) into WMMA-B layout.
__global__ void pack_cols2(const float* __restrict__ Wa, int Ka,
                           const float* __restrict__ Wb, int Kb,
                           h16* __restrict__ out, int N) {
  int Kd = Ka + Kb;
  size_t total = (size_t)N * Kd;
  size_t idx = (size_t)blockIdx.x * blockDim.x + threadIdx.x;
  if (idx >= total) return;
  int j = idx & 15;
  int lane = (idx >> 4) & 31;
  int blk = (int)(idx >> 9);
  int KC = Kd >> 5;
  int kc = blk % KC;
  int nt = blk / KC;
  int row = nt * 16 + (lane & 15);
  int col = kc * 32 + (lane >> 4) * 16 + j;
  float v = (col < Ka) ? Wa[(size_t)row * Ka + col] : Wb[(size_t)row * Kb + (col - Ka)];
  out[idx] = (h16)v;
}
// Pack row-concat [Wa;Wb], both (Na x Kd), into WMMA-B layout (N = 2*Na).
__global__ void pack_rows2(const float* __restrict__ Wa, const float* __restrict__ Wb,
                           int Na, int Kd, h16* __restrict__ out) {
  int N = 2 * Na;
  size_t total = (size_t)N * Kd;
  size_t idx = (size_t)blockIdx.x * blockDim.x + threadIdx.x;
  if (idx >= total) return;
  int j = idx & 15;
  int lane = (idx >> 4) & 31;
  int blk = (int)(idx >> 9);
  int KC = Kd >> 5;
  int kc = blk % KC;
  int nt = blk / KC;
  int row = nt * 16 + (lane & 15);
  int col = kc * 32 + (lane >> 4) * 16 + j;
  float v = (row < Na) ? Wa[(size_t)row * Kd + col] : Wb[(size_t)(row - Na) * Kd + col];
  out[idx] = (h16)v;
}
__global__ void cvt_f16(const float* __restrict__ in, h16* __restrict__ out, int n) {
  int i = blockIdx.x * blockDim.x + threadIdx.x;
  if (i < n) out[i] = (h16)in[i];
}
__global__ void copy_targets(const int* __restrict__ tgt, float* __restrict__ out) {
  int i = blockIdx.x * blockDim.x + threadIdx.x;     // i = b*200 + s
  if (i < BB * TDEC) {
    int b = i / TDEC, s = i % TDEC;
    out[i] = (float)tgt[b * LTGT + s + 1];
  }
}
__global__ void init_state(const float* __restrict__ h1_0, const float* __restrict__ c1_0,
                           const float* __restrict__ h2_0, const float* __restrict__ c2_0,
                           h16* __restrict__ X1, h16* __restrict__ X2,
                           float* __restrict__ H2, float* __restrict__ C1, float* __restrict__ C2) {
  int idx = blockIdx.x * blockDim.x + threadIdx.x;
  if (idx < 2 * BB * IN1) {
    int buf = idx / (BB * IN1);
    int r = (idx / IN1) % BB;
    int col = idx % IN1;
    float v = (buf == 1 && col >= (EE + VV)) ? h1_0[r * HH + col - (EE + VV)] : 0.f;
    X1[idx] = (h16)v;
  }
  if (idx < 2 * BB * IN2) {
    int buf = idx / (BB * IN2);
    int r = (idx / IN2) % BB;
    int col = idx % IN2;
    float v = (buf == 1 && col >= HH) ? h2_0[r * HH + col - HH] : 0.f;
    X2[idx] = (h16)v;
  }
  if (idx < BB * HH) {
    C1[idx] = c1_0[idx];
    C2[idx] = c2_0[idx];
    H2[idx] = h2_0[idx];
  }
}

// ---------------- keys/vals precompute GEMM (WMMA) ----------------
// keys[b,t,:] = listener[b,t,:] @ wh^T + bh ; vals = (listener @ wv^T + bv) * mask
// M = B*TLIS = 51200, N = 256 (keys|vals), K = 512. 800 WGs x 512 threads:
// 16 waves = 4 m-tiles x 4 n-groups; each wave does 4 n-tiles of its group.
__global__ __launch_bounds__(512) void keysvals_kernel(
    const float* __restrict__ lis, const int* __restrict__ lens,
    const float* __restrict__ bh, const float* __restrict__ bv,
    const h16* __restrict__ Wkv, h16* __restrict__ Keys, h16* __restrict__ Vals) {
  int tid = threadIdx.x, wave = tid >> 5, lane = tid & 31;
  int mi = wave & 3, ng = wave >> 2;
  int row0 = (blockIdx.x * 4 + mi) * 16;       // global row (b*TLIS + t)
  int b = row0 / TLIS;                         // 800 % 16 == 0 -> tile stays in one b
  const int KC = DLIS / 32;
  v8f acc[4];
#pragma unroll
  for (int q = 0; q < 4; ++q)
#pragma unroll
    for (int v = 0; v < 8; ++v) acc[q][v] = 0.f;

  for (int kc = 0; kc < KC; ++kc) {
    v16h a = load_a_f32(lis, DLIS, row0, kc * 32, lane);
#pragma unroll
    for (int q = 0; q < 4; ++q) {
      v16h bm = load_b_pk(Wkv, KC, ng * 4 + q, kc, lane);
      acc[q] = __builtin_amdgcn_wmma_f32_16x16x32_f16(false, a, false, bm, (short)0, acc[q], false, false);
    }
  }
  int c = lane & 15, hi = lane >> 4;
  int len = lens[b];
#pragma unroll
  for (int q = 0; q < 4; ++q) {
    int n = (ng * 4 + q) * 16 + c;
    float bias = (n < KK) ? bh[n] : bv[n - KK];
#pragma unroll
    for (int v = 0; v < 8; ++v) {
      int row = row0 + hi * 8 + v;
      int t = row - b * TLIS;
      float val = acc[q][v] + bias;
      if (n < KK) {
        Keys[((size_t)b * TLIS + t) * KK + n] = (h16)val;
      } else {
        float m = (b == 0 || t < len) ? 1.f : 0.f;
        Vals[((size_t)b * TLIS + t) * VV + (n - KK)] = (h16)(val * m);
      }
    }
  }
}

// ---------------- fused LSTM cell (gate GEMM via WMMA + nonlinearity) ----------------
// Xin: 64 x IN f16; Wpk: packed (4H x IN). Each WG handles 64 hidden units of all
// 4 gates (i,f,g,o). 32 WGs x 512 threads; 16 waves = 4 m-tiles x 4 n-subtiles.
__global__ __launch_bounds__(512) void lstm_kernel(
    const h16* __restrict__ Xin, const h16* __restrict__ Wpk,
    const float* __restrict__ bih, const float* __restrict__ bhh,
    float* __restrict__ Cst,
    h16* __restrict__ HoutA, int ldA, int offA,
    h16* __restrict__ HoutB, int ldB, int offB,
    float* __restrict__ HoutF, int IN) {
  int n0 = blockIdx.x * 64;                    // within-gate column base
  int tid = threadIdx.x, wave = tid >> 5, lane = tid & 31;
  int mi = wave & 3, ni = wave >> 2;
  int row0 = mi * 16;
  int KC = IN >> 5;
  v8f acc[4];
#pragma unroll
  for (int g = 0; g < 4; ++g)
#pragma unroll
    for (int v = 0; v < 8; ++v) acc[g][v] = 0.f;

  for (int kc = 0; kc < KC; ++kc) {
    v16h a = load_a_f16(Xin, IN, row0, kc * 32, lane);
#pragma unroll
    for (int g = 0; g < 4; ++g) {
      int nt = (g * HH + n0) / 16 + ni;
      v16h bm = load_b_pk(Wpk, KC, nt, kc, lane);
      acc[g] = __builtin_amdgcn_wmma_f32_16x16x32_f16(false, a, false, bm, (short)0, acc[g], false, false);
    }
  }
  int c = lane & 15, hi = lane >> 4;
  int col = n0 + ni * 16 + c;                  // 0..511
  float bi = bih[0 * HH + col] + bhh[0 * HH + col];
  float bf = bih[1 * HH + col] + bhh[1 * HH + col];
  float bg = bih[2 * HH + col] + bhh[2 * HH + col];
  float bo = bih[3 * HH + col] + bhh[3 * HH + col];
#pragma unroll
  for (int v = 0; v < 8; ++v) {
    int row = row0 + hi * 8 + v;
    float gi = acc[0][v] + bi;
    float gf = acc[1][v] + bf;
    float gg = acc[2][v] + bg;
    float go = acc[3][v] + bo;
    float cp = Cst[row * HH + col];
    float cn = sigf(gf) * cp + sigf(gi) * tanhf(gg);
    float hn = sigf(go) * tanhf(cn);
    Cst[row * HH + col] = cn;
    HoutA[(size_t)row * ldA + offA + col] = (h16)hn;
    if (HoutB) HoutB[(size_t)row * ldB + offB + col] = (h16)hn;
    if (HoutF) HoutF[row * HH + col] = hn;
  }
}

// ---------------- fused attention + output projection (one WG per batch row) ----------------
__global__ __launch_bounds__(256) void attend_out_kernel(
    const float* __restrict__ H2,
    const h16* __restrict__ Keys, const h16* __restrict__ Vals,
    const h16* __restrict__ Wsf, const float* __restrict__ bs,
    const h16* __restrict__ Woutf, const float* __restrict__ bout,
    const h16* __restrict__ Embf,
    const int* __restrict__ lens, const int* __restrict__ tgt,
    h16* __restrict__ X1next,
    float* __restrict__ out_probs, float* __restrict__ out_preds,
    float* __restrict__ out_att, int s) {
  int b = blockIdx.x, tid = threadIdx.x;
  __shared__ float h2s[HH];
  __shared__ float qs[KK];
  __shared__ float as[TLIS];
  __shared__ float red[256];
  __shared__ float ctx[VV];
  __shared__ float logitv[CC];

  for (int i = tid; i < HH; i += 256) h2s[i] = H2[b * HH + i];
  __syncthreads();

  if (tid < KK) {                               // q = h2 @ ws^T + bs
    const h16* w = Wsf + (size_t)tid * HH;
    float a = bs[tid];
    for (int h = 0; h < HH; ++h) a += h2s[h] * (float)w[h];
    qs[tid] = a;
  }
  __syncthreads();

  int len = lens[b];
  float ev[4];
  float lmax = -INFINITY;
#pragma unroll
  for (int k = 0; k < 4; ++k) {
    int t = tid + k * 256;
    ev[k] = -INFINITY;
    if (t < TLIS) {
      bool valid = (b == 0) || (t < len);
      if (valid) {
        const h16* kp = Keys + ((size_t)b * TLIS + t) * KK;
        float e = 0.f;
        for (int j = 0; j < KK; ++j) e += qs[j] * (float)kp[j];
        ev[k] = e;
        lmax = fmaxf(lmax, e);
      }
    }
  }
  red[tid] = lmax; __syncthreads();
  for (int off = 128; off; off >>= 1) { if (tid < off) red[tid] = fmaxf(red[tid], red[tid + off]); __syncthreads(); }
  float M = red[0]; __syncthreads();

  float lsum = 0.f;
#pragma unroll
  for (int k = 0; k < 4; ++k) {
    int t = tid + k * 256;
    if (t < TLIS) {
      float p = (ev[k] == -INFINITY) ? 0.f : __expf(ev[k] - M);
      as[t] = p;
      lsum += p;
    }
  }
  red[tid] = lsum; __syncthreads();
  for (int off = 128; off; off >>= 1) { if (tid < off) red[tid] += red[tid + off]; __syncthreads(); }
  float inv = 1.f / fmaxf(red[0], 1e-12f); __syncthreads();

#pragma unroll
  for (int k = 0; k < 4; ++k) {
    int t = tid + k * 256;
    if (t < TLIS) {
      float a = as[t] * inv;
      as[t] = a;
      out_att[((size_t)b * TDEC + s) * TLIS + t] = a;
    }
  }
  __syncthreads();

  // ctx[v] = sum_t a[t] * vals[b,t,v]   (two half-range partials)
  {
    int g = tid >> 7, v = tid & 127;
    float part = 0.f;
    for (int t = g * 400; t < (g + 1) * 400; ++t)
      part += as[t] * (float)Vals[((size_t)b * TLIS + t) * VV + v];
    red[tid] = part; __syncthreads();
    if (tid < VV) ctx[tid] = red[tid] + red[tid + 128];
    __syncthreads();
  }

  if (tid < CC) {                               // logits = [h2|ctx] @ w_out^T + b_out
    const h16* w = Woutf + (size_t)tid * (HH + VV);
    float a = bout[tid];
    for (int h = 0; h < HH; ++h) a += h2s[h] * (float)w[h];
    for (int j = 0; j < VV; ++j) a += ctx[j] * (float)w[HH + j];
    logitv[tid] = a;
    out_probs[((size_t)b * CC + tid) * TDEC + s] = a;
  }
  __syncthreads();
  if (tid == 0) {
    int bi = 0; float bv = logitv[0];
    for (int c = 1; c < CC; ++c) if (logitv[c] > bv) { bv = logitv[c]; bi = c; }
    out_preds[b * TDEC + s] = (float)bi;
  }
  // stage next-step x = [emb[tok], ctx] into X1 (next parity)
  int tok = tgt[b * LTGT + s];
  if (tid < EE) X1next[(size_t)b * IN1 + tid] = Embf[(size_t)tok * EE + tid];
  if (tid < VV) X1next[(size_t)b * IN1 + EE + tid] = (h16)ctx[tid];
}

// ---------------- host side ----------------
static inline int cdiv(size_t a, int b) { return (int)((a + b - 1) / b); }

extern "C" void kernel_launch(void* const* d_in, const int* in_sizes, int n_in,
                              void* d_out, int out_size, void* d_ws, size_t ws_size,
                              hipStream_t stream) {
  const float* listener = (const float*)d_in[0];
  const int*   lens     = (const int*)d_in[1];
  const int*   tgt      = (const int*)d_in[2];
  const float* emb      = (const float*)d_in[3];
  const float* w_ih1    = (const float*)d_in[4];
  const float* w_hh1    = (const float*)d_in[5];
  const float* b_ih1    = (const float*)d_in[6];
  const float* b_hh1    = (const float*)d_in[7];
  const float* w_ih2    = (const float*)d_in[8];
  const float* w_hh2    = (const float*)d_in[9];
  const float* b_ih2    = (const float*)d_in[10];
  const float* b_hh2    = (const float*)d_in[11];
  const float* ws_w     = (const float*)d_in[12];
  const float* bs       = (const float*)d_in[13];
  const float* wh       = (const float*)d_in[14];
  const float* bh       = (const float*)d_in[15];
  const float* wv       = (const float*)d_in[16];
  const float* bv       = (const float*)d_in[17];
  const float* w_out    = (const float*)d_in[18];
  const float* b_out    = (const float*)d_in[19];
  const float* h1_0     = (const float*)d_in[20];
  const float* c1_0     = (const float*)d_in[21];
  const float* h2_0     = (const float*)d_in[22];
  const float* c2_0     = (const float*)d_in[23];

  // workspace layout (all offsets 256B-aligned)
  char* w = (char*)d_ws;
  size_t off = 0;
  h16* WC1  = (h16*)(w + off); off += (size_t)G4H * IN1 * 2;          // 3,670,016
  h16* WC2  = (h16*)(w + off); off += (size_t)G4H * IN2 * 2;          // 4,194,304
  h16* WKV  = (h16*)(w + off); off += (size_t)256 * DLIS * 2;         //   262,144
  h16* WS16 = (h16*)(w + off); off += (size_t)KK * HH * 2;            //   131,072
  h16* WO16 = (h16*)(w + off); off += (size_t)CC * (HH + VV) * 2;     //    81,920
  h16* EMB16= (h16*)(w + off); off += (size_t)CC * EE * 2;            //    32,768
  h16* Keys = (h16*)(w + off); off += (size_t)BB * TLIS * KK * 2;     // 13,107,200
  h16* Vals = (h16*)(w + off); off += (size_t)BB * TLIS * VV * 2;     // 13,107,200
  h16* X1   = (h16*)(w + off); off += (size_t)2 * BB * IN1 * 2;       //   229,376
  h16* X2   = (h16*)(w + off); off += (size_t)2 * BB * IN2 * 2;       //   262,144
  float* H2 = (float*)(w + off); off += (size_t)BB * HH * 4;          //   131,072
  float* C1 = (float*)(w + off); off += (size_t)BB * HH * 4;
  float* C2 = (float*)(w + off); off += (size_t)BB * HH * 4;
  if (off > ws_size) return;   // insufficient scratch: deterministic no-op

  float* out_probs = (float*)d_out;                                   // (B,C,T)
  float* out_preds = out_probs + (size_t)BB * CC * TDEC;              // (B,T)
  float* out_tgts  = out_preds + (size_t)BB * TDEC;                   // (B,T)
  float* out_att   = out_tgts  + (size_t)BB * TDEC;                   // (B,T,TLIS)

  // --- one-time prep ---
  pack_cols2<<<cdiv((size_t)G4H * IN1, 256), 256, 0, stream>>>(w_ih1, EE + VV, w_hh1, HH, WC1, G4H);
  pack_cols2<<<cdiv((size_t)G4H * IN2, 256), 256, 0, stream>>>(w_ih2, HH, w_hh2, HH, WC2, G4H);
  pack_rows2<<<cdiv((size_t)256 * DLIS, 256), 256, 0, stream>>>(wh, wv, KK, DLIS, WKV);
  cvt_f16<<<cdiv(KK * HH, 256), 256, 0, stream>>>(ws_w, WS16, KK * HH);
  cvt_f16<<<cdiv(CC * (HH + VV), 256), 256, 0, stream>>>(w_out, WO16, CC * (HH + VV));
  cvt_f16<<<cdiv(CC * EE, 256), 256, 0, stream>>>(emb, EMB16, CC * EE);
  init_state<<<cdiv(2 * BB * IN2, 256), 256, 0, stream>>>(h1_0, c1_0, h2_0, c2_0, X1, X2, H2, C1, C2);
  copy_targets<<<cdiv(BB * TDEC, 256), 256, 0, stream>>>(tgt, out_tgts);

  // --- keys/vals precompute: 3200 m-tiles / 4 per WG = 800 WGs ---
  keysvals_kernel<<<800, 512, 0, stream>>>(listener, lens, bh, bv, WKV, Keys, Vals);

  // --- sequential decode: step 0 = attend/output with h2_0; steps 1..199 full ---
  for (int s = 0; s < TDEC; ++s) {
    int p = s & 1;
    if (s >= 1) {
      // LSTM1: reads X1[p] = [x_s | h1_{s-1}]; writes h1_s -> X2[p][:,:512] and X1[1-p][:,384:]
      lstm_kernel<<<G4H / 64, 512, 0, stream>>>(
          X1 + (size_t)p * BB * IN1, WC1, b_ih1, b_hh1, C1,
          X2 + (size_t)p * BB * IN2, IN2, 0,
          X1 + (size_t)(1 - p) * BB * IN1, IN1, EE + VV,
          (float*)nullptr, IN1);
      // LSTM2: reads X2[p] = [h1_s | h2_{s-1}]; writes h2_s -> X2[1-p][:,512:] and H2 (f32)
      lstm_kernel<<<G4H / 64, 512, 0, stream>>>(
          X2 + (size_t)p * BB * IN2, WC2, b_ih2, b_hh2, C2,
          X2 + (size_t)(1 - p) * BB * IN2, IN2, HH,
          (h16*)nullptr, 0, 0,
          H2, IN2);
    }
    // attention + output; also stages x_{s+1} = [emb[tok_s], ctx_s] into X1[1-p]
    attend_out_kernel<<<BB, 256, 0, stream>>>(
        H2, Keys, Vals, WS16, bs, WO16, b_out, EMB16, lens, tgt,
        X1 + (size_t)(1 - p) * BB * IN1,
        out_probs, out_preds, out_att, s);
  }
}
